// MaskedSelfAttention_25451976196343
// MI455X (gfx1250) — compile-verified, matrix operand order checked by reading, not run on hardware
//
#include <hip/hip_runtime.h>

#define S_DIM 8192
#define D_DIM 256
#define SCALE_F 0.0625f   // 1/sqrt(256)

typedef __attribute__((ext_vector_type(16))) _Float16 v16h;
typedef __attribute__((ext_vector_type(8)))  _Float16 v8h;
typedef __attribute__((ext_vector_type(8)))  float    v8f;

__device__ __forceinline__ v8f wmma_f16(v16h a, v16h b, v8f c) {
  // 8 args: (neg_a, A, neg_b, B, c_mod, C, reuse_a, reuse_b)
  return __builtin_amdgcn_wmma_f32_16x16x32_f16(false, a, false, b, (short)0, c,
                                                false, false);
}

// Load a 16x32 f16 operand tile in the WMMA A/B register layout
// (CDNA5 ISA 7.12.2): lane l holds row (l&15); lanes 0-15 take
// K {0..7,16..23}, lanes 16-31 take K {8..15,24..31}. Two contiguous
// 16-byte loads per lane. `stride` = row stride in elements.
__device__ __forceinline__ v16h load_tile_f16(const _Float16* base, int stride,
                                              int lane) {
  const int m  = lane & 15;
  const int hi = (lane >> 4) & 1;
  const _Float16* p = base + m * stride + hi * 8;
  v8h lo = *(const v8h*)(p);
  v8h hv = *(const v8h*)(p + 16);
  v16h r;
#pragma unroll
  for (int i = 0; i < 8; ++i) { r[i] = lo[i]; r[i + 8] = hv[i]; }
  return r;
}

// Same layout but sourced from fp32 memory with conversion to f16.
__device__ __forceinline__ v16h load_tile_f32cvt(const float* base, int stride,
                                                 int lane) {
  const int m  = lane & 15;
  const int hi = (lane >> 4) & 1;
  const float* p = base + m * stride + hi * 8;
  v16h r;
#pragma unroll
  for (int i = 0; i < 8; ++i) {
    r[i]     = (_Float16)p[i];
    r[i + 8] = (_Float16)p[16 + i];
  }
  return r;
}

// ---------------------------------------------------------------------------
// Y = X @ W.T + b, output f16. X:[S,256] f32, W:[256,256] f32 (torch Linear).
// TRANSPOSE_OUT=1 stores Y transposed ([256][S]) for the V matrix so the
// PV-stage B operand columns are contiguous.
// One wave per 16x16 output tile; 4 waves per block.
// ---------------------------------------------------------------------------
template <int TRANSPOSE_OUT>
__global__ __launch_bounds__(128) void proj_kernel(
    const float* __restrict__ X, const float* __restrict__ W,
    const float* __restrict__ bias, _Float16* __restrict__ Y) {
  const int lane = threadIdx.x & 31;
  const int gw   = blockIdx.x * 4 + (threadIdx.x >> 5);
  const int row0 = (gw >> 4) * 16;   // 512 row tiles
  const int col0 = (gw & 15) * 16;   // 16 col tiles
  v8f acc = {};
#pragma unroll
  for (int kk = 0; kk < D_DIM; kk += 32) {
    v16h a = load_tile_f32cvt(X + row0 * D_DIM + kk, D_DIM, lane);
    v16h b = load_tile_f32cvt(W + col0 * D_DIM + kk, D_DIM, lane);  // B col n = W row n
    acc = wmma_f16(a, b, acc);
  }
  const int n  = col0 + (lane & 15);
  const int hi = (lane >> 4) & 1;
  const float bn = bias[n];
#pragma unroll
  for (int r = 0; r < 8; ++r) {
    const int row = row0 + r + hi * 8;   // C/D layout: VGPR r -> row r (+8 for hi half)
    const float v = acc[r] + bn;
    if (TRANSPOSE_OUT) Y[(size_t)n * S_DIM + row] = (_Float16)v;
    else               Y[(size_t)row * D_DIM + n] = (_Float16)v;
  }
}

// ---------------------------------------------------------------------------
// Fused flash attention with multiplicative mask before softmax.
// Each wave: 16 rows, streams all 8192 columns in steps of 32.
// Streams the 268MB fp32 mask exactly once (HBM-bound lower limit ~11.5us).
// Steady-state fast path skips the accumulator rescale (alpha==1) whenever
// no running row-max changed; branch is made wave-uniform via ballot.
// PV B-tiles are manually software-pipelined 4 deep to keep ~8 global_load
// b128s in flight instead of stalling on s_wait_loadcnt 0 per WMMA.
// ---------------------------------------------------------------------------
__global__ __launch_bounds__(128) void attn_kernel(
    const _Float16* __restrict__ qh, const _Float16* __restrict__ kh,
    const _Float16* __restrict__ vT, const float* __restrict__ mask,
    _Float16* __restrict__ hOut) {
  __shared__ _Float16 pstage[4][16 * 32];  // 1KB per wave: P tile repack buffer
  const int lane = threadIdx.x & 31;
  const int w    = threadIdx.x >> 5;
  const int row0 = (blockIdx.x * 4 + w) * 16;
  const int m16  = lane & 15;
  const int hi   = (lane >> 4) & 1;
  _Float16* ps = &pstage[w][0];

  v8f zero = {};
  v8f acc[16];                 // h accumulator: 16 N-tiles x 8 rows (fp32)
#pragma unroll
  for (int j = 0; j < 16; ++j) acc[j] = zero;
  float mrun[8], lrun[8];      // per-row running max / sum (rows r + hi*8)
#pragma unroll
  for (int r = 0; r < 8; ++r) { mrun[r] = -1.0e30f; lrun[r] = 0.0f; }

  for (int n0 = 0; n0 < S_DIM; n0 += 32) {
    // ---- scores: S[16x32] = q_tile(16x256) @ k(n0..n0+31,:)^T
    v8f s0 = zero, s1 = zero;
#pragma unroll
    for (int kk = 0; kk < D_DIM; kk += 32) {
      v16h a  = load_tile_f16(qh + row0 * D_DIM + kk, D_DIM, lane);
      v16h b0 = load_tile_f16(kh + n0 * D_DIM + kk, D_DIM, lane);
      v16h b1 = load_tile_f16(kh + (n0 + 16) * D_DIM + kk, D_DIM, lane);
      s0 = wmma_f16(a, b0, s0);
      s1 = wmma_f16(a, b1, s1);
    }
    // hint the next mask tile into cache while we do softmax + PV
    if (n0 + 32 < S_DIM)
      __builtin_prefetch(&mask[(size_t)(row0 + m16) * S_DIM + n0 + 32 + hi * 16], 0, 0);

    // ---- multiplicative mask + scale (masked BEFORE softmax)
    float t[16];
#pragma unroll
    for (int r = 0; r < 8; ++r) {
      const size_t mrow = (size_t)(row0 + r + hi * 8) * S_DIM;
      t[r]     = s0[r] * mask[mrow + n0 + m16]      * SCALE_F;
      t[8 + r] = s1[r] * mask[mrow + n0 + 16 + m16] * SCALE_F;
    }

    // ---- per-row block max; rows live in 16-lane halves -> shfl_xor ladder
    float mx[8];
    bool upd = false;
#pragma unroll
    for (int r = 0; r < 8; ++r) {
      float v = fmaxf(t[r], t[8 + r]);
#pragma unroll
      for (int off = 1; off < 16; off <<= 1)
        v = fmaxf(v, __shfl_xor(v, off, 32));
      mx[r] = v;
      upd = upd || (v > mrun[r]);
    }
    // wave-uniform slow path: only rescale when some running max increased
    if (__builtin_amdgcn_ballot_w32(upd)) {
#pragma unroll
      for (int r = 0; r < 8; ++r) {
        const float mn    = fmaxf(mrun[r], mx[r]);
        const float alpha = __expf(mrun[r] - mn);
        mrun[r] = mn;
        lrun[r] *= alpha;
#pragma unroll
        for (int j = 0; j < 16; ++j) acc[j][r] *= alpha;
      }
    }
    // ---- probabilities + running denominator; stage P for the PV matmul
#pragma unroll
    for (int r = 0; r < 8; ++r) {
      const float p0 = __expf(t[r] - mrun[r]);
      const float p1 = __expf(t[8 + r] - mrun[r]);
      float rs = p0 + p1;
#pragma unroll
      for (int off = 1; off < 16; off <<= 1)
        rs += __shfl_xor(rs, off, 32);
      lrun[r] += rs;
      // repack P from C-layout to row-major staging (row r+hi*8)
      ps[(r + hi * 8) * 32 + m16]      = (_Float16)p0;
      ps[(r + hi * 8) * 32 + 16 + m16] = (_Float16)p1;
    }
    asm volatile("s_wait_dscnt 0" ::: "memory");  // same-wave LDS RAW guard
    v16h pa = load_tile_f16(ps, 32, lane);        // P tile in A layout

    // ---- PV: acc(16x256) += P(16x32) @ V(n0..n0+31, 0..255); B from vT rows
    // 4-deep rotating buffer: keep 4 tile-loads (8x global_load_b128) in
    // flight so each WMMA waits on a load issued 4 iterations earlier.
    v16h bb[4];
#pragma unroll
    for (int j = 0; j < 4; ++j)
      bb[j] = load_tile_f16(vT + (size_t)(j * 16) * S_DIM + n0, S_DIM, lane);
#pragma unroll
    for (int j = 0; j < 16; ++j) {
      v16h bcur = bb[j & 3];
      if (j + 4 < 16)
        bb[j & 3] = load_tile_f16(vT + (size_t)((j + 4) * 16) * S_DIM + n0,
                                  S_DIM, lane);
      acc[j] = wmma_f16(pa, bcur, acc[j]);
    }
  }

  // ---- normalize by the softmax denominator and emit h as f16
#pragma unroll
  for (int r = 0; r < 8; ++r) {
    const float inv = 1.0f / lrun[r];
    const int row = row0 + r + hi * 8;
#pragma unroll
    for (int j = 0; j < 16; ++j)
      hOut[(size_t)row * D_DIM + j * 16 + m16] = (_Float16)(acc[j][r] * inv);
  }
}

// ---------------------------------------------------------------------------
// out = h @ Wo.T + bo, fp32 output.
// ---------------------------------------------------------------------------
__global__ __launch_bounds__(128) void out_kernel(
    const _Float16* __restrict__ H, const float* __restrict__ Wo,
    const float* __restrict__ bo, float* __restrict__ out) {
  const int lane = threadIdx.x & 31;
  const int gw   = blockIdx.x * 4 + (threadIdx.x >> 5);
  const int row0 = (gw >> 4) * 16;
  const int col0 = (gw & 15) * 16;
  v8f acc = {};
#pragma unroll
  for (int kk = 0; kk < D_DIM; kk += 32) {
    v16h a = load_tile_f16(H + row0 * D_DIM + kk, D_DIM, lane);
    v16h b = load_tile_f32cvt(Wo + col0 * D_DIM + kk, D_DIM, lane);
    acc = wmma_f16(a, b, acc);
  }
  const int n  = col0 + (lane & 15);
  const int hi = (lane >> 4) & 1;
  const float bn = bo[n];
#pragma unroll
  for (int r = 0; r < 8; ++r)
    out[(size_t)(row0 + r + hi * 8) * D_DIM + n] = acc[r] + bn;
}

extern "C" void kernel_launch(void* const* d_in, const int* in_sizes, int n_in,
                              void* d_out, int out_size, void* d_ws, size_t ws_size,
                              hipStream_t stream) {
  const float* query = (const float*)d_in[0];
  const float* key_  = (const float*)d_in[1];
  const float* value = (const float*)d_in[2];
  const float* mask  = (const float*)d_in[3];
  const float* Wq    = (const float*)d_in[4];
  const float* bq    = (const float*)d_in[5];
  const float* Wk    = (const float*)d_in[6];
  const float* bk    = (const float*)d_in[7];
  const float* Wv    = (const float*)d_in[8];
  const float* bv    = (const float*)d_in[9];
  const float* Wo    = (const float*)d_in[10];
  const float* bo    = (const float*)d_in[11];

  char* ws = (char*)d_ws;
  const size_t bytesM = (size_t)S_DIM * D_DIM * sizeof(_Float16);  // 4MB each
  _Float16* qh = (_Float16*)(ws);
  _Float16* kh = (_Float16*)(ws + bytesM);
  _Float16* vT = (_Float16*)(ws + 2 * bytesM);   // stored transposed [256][S]
  _Float16* hh = (_Float16*)(ws + 3 * bytesM);

  dim3 blk(128);                                        // 4 waves (wave32)
  dim3 gProj((S_DIM / 16) * (D_DIM / 16) / 4);          // 2048 blocks

  hipLaunchKernelGGL((proj_kernel<0>), gProj, blk, 0, stream, query, Wq, bq, qh);
  hipLaunchKernelGGL((proj_kernel<0>), gProj, blk, 0, stream, key_,  Wk, bk, kh);
  hipLaunchKernelGGL((proj_kernel<1>), gProj, blk, 0, stream, value, Wv, bv, vT);

  dim3 gAttn((S_DIM / 16) / 4);                         // 128 blocks, wave=16 rows
  hipLaunchKernelGGL(attn_kernel, gAttn, blk, 0, stream, qh, kh, vT, mask, hh);

  hipLaunchKernelGGL(out_kernel, gProj, blk, 0, stream, hh, Wo, bo, (float*)d_out);
}
